// MLPNetOWT_BN_26018911879634
// MI455X (gfx1250) — 8-bit matrix operand order repaired and checked by reading, compiled, not run on hardware
//
#include <hip/hip_runtime.h>
#include <stdint.h>

typedef __attribute__((ext_vector_type(16))) int   v16i;
typedef __attribute__((ext_vector_type(8)))  float v8f;

#define FP8_P1 0x38u
#define FP8_M1 0xB8u
#define EPSV   1e-5f

#define BM 128
#define BN 128
#define BK 128

__device__ __forceinline__ uint32_t sgn8(float x) { return x > 0.0f ? FP8_P1 : FP8_M1; }

__device__ __forceinline__ uint32_t lds_off32(const void* p) {
    return (uint32_t)(uintptr_t)p;
}

__device__ __forceinline__ void async_copy_b128(uint32_t lds, const uint8_t* g) {
    asm volatile("global_load_async_to_lds_b128 %0, %1, off"
                 :: "v"(lds), "v"((uint64_t)(uintptr_t)g)
                 : "memory");
}

__global__ __launch_bounds__(256)
void k_binarize_act(const float* __restrict__ x, uint32_t* __restrict__ q, int n4) {
    int i = blockIdx.x * 256 + threadIdx.x;
    if (i >= n4) return;
    float4 f = ((const float4*)x)[i];
    q[i] = sgn8(f.x) | (sgn8(f.y) << 8) | (sgn8(f.z) << 16) | (sgn8(f.w) << 24);
}

__global__ __launch_bounds__(256)
void k_binarize_wgt(const float* __restrict__ W, uint32_t* __restrict__ q,
                    int rows, int k4PerRow) {
    int i = blockIdx.x * 256 + threadIdx.x;
    int row = i / k4PerRow;
    uint32_t w = 0;
    if (row < rows) {
        float4 f = ((const float4*)W)[(size_t)i];
        w = sgn8(f.x) | (sgn8(f.y) << 8) | (sgn8(f.z) << 16) | (sgn8(f.w) << 24);
    }
    q[i] = w;
}

__global__ __launch_bounds__(256)
void k_bgemm_fp8(const uint8_t* __restrict__ A,
                 const uint8_t* __restrict__ B,
                 int K, int mode,
                 const float* __restrict__ bias,
                 const float* __restrict__ gam,
                 const float* __restrict__ bet,
                 const float* __restrict__ mu,
                 const float* __restrict__ var,
                 uint8_t* __restrict__ outq, int outq_stride,
                 float* __restrict__ outf, int outf_cols)
{
    __shared__ __align__(16) uint8_t As[2][BM * BK];
    __shared__ __align__(16) uint8_t Bs[2][BN * BK];

    const int tid  = threadIdx.x;
    const int lane = tid & 31;
    const int wave = tid >> 5;
    const int wr   = wave & 3;
    const int wc   = wave >> 2;
    const int half = lane >> 4;
    const int l16  = lane & 15;

    const size_t Abase = (size_t)blockIdx.x * BM * (size_t)K;
    const size_t Bbase = (size_t)blockIdx.y * BN * (size_t)K;

    const int colB = (tid & 7) << 4;
    const int row0 = tid >> 3;
    const uint8_t* gA = A + Abase + (size_t)row0 * K + colB;
    const uint8_t* gB = B + Bbase + (size_t)row0 * K + colB;
    const uint32_t lA0 = lds_off32(&As[0][row0 * BK + colB]);
    const uint32_t lB0 = lds_off32(&Bs[0][row0 * BK + colB]);
    const uint32_t stageBytes = (uint32_t)(BM * BK);

    auto issue_tile = [&](int s, int k0) {
        const uint32_t so = (uint32_t)s * stageBytes;
#pragma unroll
        for (int c = 0; c < 4; ++c) {
            async_copy_b128(lA0 + so + (uint32_t)(c * 32 * BK),
                            gA + (size_t)k0 + (size_t)(c * 32) * K);
            async_copy_b128(lB0 + so + (uint32_t)(c * 32 * BK),
                            gB + (size_t)k0 + (size_t)(c * 32) * K);
        }
    };

    v8f zero = {};
    v8f acc[2][4];
#pragma unroll
    for (int i = 0; i < 2; ++i)
#pragma unroll
        for (int j = 0; j < 4; ++j) acc[i][j] = zero;

    const int nsteps = K / BK;
    issue_tile(0, 0);

    for (int s = 0; s < nsteps; ++s) {
        const int cur = s & 1;
        if (s + 1 < nsteps) {
            issue_tile(cur ^ 1, (s + 1) * BK);
            asm volatile("s_wait_asynccnt 0x8" ::: "memory");
        } else {
            asm volatile("s_wait_asynccnt 0x0" ::: "memory");
        }
        __syncthreads();

        v16i af[2];
#pragma unroll
        for (int i = 0; i < 2; ++i) {
            const uint8_t* ap = &As[cur][(wr * 32 + i * 16 + l16) * BK + half * 16];
#pragma unroll
            for (int g = 0; g < 4; ++g) {
                uint4 d = *(const uint4*)(ap + g * 32);
                af[i][4 * g]     = (int)d.x;
                af[i][4 * g + 1] = (int)d.y;
                af[i][4 * g + 2] = (int)d.z;
                af[i][4 * g + 3] = (int)d.w;
            }
        }
        v16i bf[4];
#pragma unroll
        for (int j = 0; j < 4; ++j) {
            const uint8_t* bp = &Bs[cur][(wc * 64 + j * 16 + l16) * BK + half * 16];
#pragma unroll
            for (int g = 0; g < 4; ++g) {
                uint4 d = *(const uint4*)(bp + g * 32);
                bf[j][4 * g]     = (int)d.x;
                bf[j][4 * g + 1] = (int)d.y;
                bf[j][4 * g + 2] = (int)d.z;
                bf[j][4 * g + 3] = (int)d.w;
            }
        }
#pragma unroll
        for (int i = 0; i < 2; ++i)
#pragma unroll
            for (int j = 0; j < 4; ++j)
                acc[i][j] = __builtin_amdgcn_wmma_f32_16x16x128_fp8_fp8(
                    af[i], bf[j], (short)0, acc[i][j], false, false);

        __syncthreads();
    }

    const int m_hi = half * 8;
#pragma unroll
    for (int j = 0; j < 4; ++j) {
        const int n_g = blockIdx.y * BN + wc * 64 + j * 16 + l16;
        if (mode == 0) {
            const float sc = gam[n_g] * rsqrtf(var[n_g] + EPSV);
            const float sh = (bias[n_g] - mu[n_g]) * sc + bet[n_g];
#pragma unroll
            for (int i = 0; i < 2; ++i) {
                const int m_base = blockIdx.x * BM + wr * 32 + i * 16 + m_hi;
#pragma unroll
                for (int r = 0; r < 8; ++r) {
                    float z = acc[i][j][r] * sc + sh;
                    outq[(size_t)(m_base + r) * outq_stride + n_g] =
                        (uint8_t)(z > 0.0f ? FP8_P1 : FP8_M1);
                }
            }
        } else if (n_g < outf_cols) {
            const float bb = bias[n_g];
#pragma unroll
            for (int i = 0; i < 2; ++i) {
                const int m_base = blockIdx.x * BM + wr * 32 + i * 16 + m_hi;
#pragma unroll
                for (int r = 0; r < 8; ++r)
                    outf[(size_t)(m_base + r) * outf_cols + n_g] = acc[i][j][r] + bb;
            }
        }
    }
}

extern "C" void kernel_launch(void* const* d_in, const int* in_sizes, int n_in,
                              void* d_out, int out_size, void* d_ws, size_t ws_size,
                              hipStream_t stream) {
    (void)in_sizes; (void)n_in; (void)out_size; (void)ws_size;

    const float* x   = (const float*)d_in[0];
    const float* W0  = (const float*)d_in[1];
    const float* b0  = (const float*)d_in[2];
    const float* g0  = (const float*)d_in[3];
    const float* be0 = (const float*)d_in[4];
    const float* m0  = (const float*)d_in[5];
    const float* v0  = (const float*)d_in[6];
    const float* W1  = (const float*)d_in[7];
    const float* b1  = (const float*)d_in[8];
    const float* g1  = (const float*)d_in[9];
    const float* be1 = (const float*)d_in[10];
    const float* m1  = (const float*)d_in[11];
    const float* v1  = (const float*)d_in[12];
    const float* W2  = (const float*)d_in[13];
    const float* b2  = (const float*)d_in[14];
    const float* g2  = (const float*)d_in[15];
    const float* be2 = (const float*)d_in[16];
    const float* m2  = (const float*)d_in[17];
    const float* v2  = (const float*)d_in[18];
    const float* W3  = (const float*)d_in[19];
    const float* b3  = (const float*)d_in[20];

    const int Bm = 8192, K0 = 2048, H = 4096, NOUT = 1000, NPAD = 1024;

    uint8_t* actA = (uint8_t*)d_ws;
    uint8_t* actB = actA + (size_t)Bm * H;
    uint8_t* wq   = actB + (size_t)Bm * H;

    {
        int n4 = Bm * K0 / 4;
        k_binarize_act<<<n4 / 256, 256, 0, stream>>>(x, (uint32_t*)actA, n4);
    }
    k_binarize_wgt<<<(H * (K0 / 4)) / 256, 256, 0, stream>>>(W0, (uint32_t*)wq, H, K0 / 4);
    {
        dim3 g(Bm / BM, H / BN);
        k_bgemm_fp8<<<g, 256, 0, stream>>>(actA, wq, K0, 0, b0, g0, be0, m0, v0,
                                           actB, H, nullptr, 0);
    }
    k_binarize_wgt<<<(H * (H / 4)) / 256, 256, 0, stream>>>(W1, (uint32_t*)wq, H, H / 4);
    {
        dim3 g(Bm / BM, H / BN);
        k_bgemm_fp8<<<g, 256, 0, stream>>>(actB, wq, H, 0, b1, g1, be1, m1, v1,
                                           actA, H, nullptr, 0);
    }
    k_binarize_wgt<<<(H * (H / 4)) / 256, 256, 0, stream>>>(W2, (uint32_t*)wq, H, H / 4);
    {
        dim3 g(Bm / BM, H / BN);
        k_bgemm_fp8<<<g, 256, 0, stream>>>(actA, wq, H, 0, b2, g2, be2, m2, v2,
                                           actB, H, nullptr, 0);
    }
    k_binarize_wgt<<<(NPAD * (H / 4)) / 256, 256, 0, stream>>>(W3, (uint32_t*)wq, NOUT, H / 4);
    {
        dim3 g(Bm / BM, NPAD / BN);
        k_bgemm_fp8<<<g, 256, 0, stream>>>(actB, wq, H, 1, b3, nullptr, nullptr, nullptr, nullptr,
                                           nullptr, 0, (float*)d_out, NOUT);
    }
}
